// SA_74775380623732
// MI455X (gfx1250) — matrix operand order repaired and checked by reading, not compiled, not run on hardware
//
#include <hip/hip_runtime.h>
#include <hip/hip_bf16.h>
#include <math.h>

#define N_ 4
#define D_ 256
#define P_ 4096

typedef __attribute__((ext_vector_type(16))) __bf16 v16bf;
typedef __attribute__((ext_vector_type(8)))  float  v8f;

__device__ __forceinline__ __bf16 f2bf(float x) { return (__bf16)x; }

// ---------------------------------------------------------------------------
// Phase 1: projections K/Q/V = W*X + b with bf16 WMMA (fp32 accumulate),
// writing results directly into WMMA-fragment-major bf16 layouts:
//   Kfrag[n][dc(d/32)][it(p/16)][lane 0..31][e 0..15]   (A-operand layout, M=p, K=d)
//   Qfrag[n][dc(d/32)][jt(p/16)][lane 0..31][e 0..15]   (B-operand layout, K=d, N=p)
//   Vfrag[n][ic(p/32)][mt(d/16)][lane 0..31][e 0..15]   (A-operand layout, M=d, K=p)
// so phase 2 fetches each operand as one contiguous 32B load per lane.
// ---------------------------------------------------------------------------
__global__ __launch_bounds__(256) void proj_frag_kernel(
    const float* __restrict__ x,                       // [N,D,P]
    const float* __restrict__ Wk, const float* __restrict__ bk,
    const float* __restrict__ Wq, const float* __restrict__ bq,
    const float* __restrict__ Wv, const float* __restrict__ bv,
    __bf16* __restrict__ Kfrag, __bf16* __restrict__ Qfrag,
    __bf16* __restrict__ Vfrag)
{
    int blk = blockIdx.x;
    int pg  = blk & 31;  blk >>= 5;    // 32 p-groups of 128
    int ot  = blk & 15;  blk >>= 4;    // 16 o-tiles of 16
    int n   = blk & 3;   blk >>= 2;    // batch
    int sel = blk;                      // 0=K 1=Q 2=V

    const float* W = (sel == 0) ? Wk : (sel == 1) ? Wq : Wv;
    const float* b = (sel == 0) ? bk : (sel == 1) ? bq : bv;

    int tid  = threadIdx.x;
    int lane = tid & 31;
    int wv   = tid >> 5;                // 8 waves, each owns one 16-wide p-tile
    int p0   = pg * 128 + wv * 16;
    int m    = lane & 15;
    int hi   = lane >> 4;

    v8f acc = {0.f, 0.f, 0.f, 0.f, 0.f, 0.f, 0.f, 0.f};

    for (int dc = 0; dc < 8; ++dc) {   // K-dim: 256 in chunks of 32
        v16bf a, bb;
        // A fragment: W[ot*16+m][dc*32 + kidx], kidx = hi*16 + e
        const float* wrow = W + (size_t)(ot * 16 + m) * D_ + dc * 32;
#pragma unroll
        for (int e = 0; e < 16; ++e) {
            int g = e >> 3, kk = e & 7;
            a[e] = f2bf(wrow[hi * 16 + e]);
        }
        // B fragment: x[n][dc*32 + hi*16 + e][p0 + m]
        const float* xcol = x + ((size_t)n * D_ + dc * 32 + hi * 16) * P_ + p0 + m;
#pragma unroll
        for (int e = 0; e < 16; ++e)
            bb[e] = f2bf(xcol[(size_t)e * P_]);

        acc = __builtin_amdgcn_wmma_f32_16x16x32_bf16(
            false, a, false, bb, (short)0, acc, false, false);
    }

    // bias + scatter into fragment layout
#pragma unroll
    for (int r = 0; r < 8; ++r) {
        int o_abs = ot * 16 + r + hi * 8;   // C layout: row = r + 8*hi
        int p_abs = p0 + m;                 // C layout: col = lane%16
        float val = acc[r] + b[o_abs];
        __bf16 bv16 = f2bf(val);
        if (sel == 0) {          // K: A-layout, M=p, K=o(d)
            int dc2 = o_abs >> 5, kidx = o_abs & 31;
            int it  = p_abs >> 4;
            int l2  = m + (((kidx & 15) >= 8) ? 16 : 0);
            int e2  = (kidx >> 4) * 8 + (kidx & 7);
            Kfrag[((((size_t)n * 8 + dc2) * 256 + it) * 32 + l2) * 16 + e2] = bv16;
        } else if (sel == 1) {   // Q: B-layout, K=o(d), N=p
            int dc2 = o_abs >> 5, kidx = o_abs & 31;
            int jt  = p_abs >> 4;
            int l2  = m + (((kidx & 15) >= 8) ? 16 : 0);
            int e2  = (kidx >> 4) * 8 + (kidx & 7);
            Qfrag[((((size_t)n * 8 + dc2) * 256 + jt) * 32 + l2) * 16 + e2] = bv16;
        } else {                 // V: A-layout, M=o(d), K=p
            int ic = p_abs >> 5, kidx = p_abs & 31;
            int m2 = r + hi * 8;
            int l2 = m2 + (((kidx & 15) >= 8) ? 16 : 0);
            int e2 = (kidx >> 4) * 8 + (kidx & 7);
            Vfrag[((((size_t)n * 128 + ic) * 16 + ot) * 32 + l2) * 16 + e2] = bv16;
        }
    }
}

// ---------------------------------------------------------------------------
// Phase 2: flash attention. One block = (batch n, 64-query tile). 8 waves.
//  - Q fragments staged to LDS with GLOBAL_LOAD_ASYNC_TO_LDS_B128 (ASYNCcnt)
//  - S-tile pass: 16 S tiles (4 i-tiles x 4 j-tiles), 2 per wave, 8 WMMA each
//  - online softmax over key axis per query column (stats in LDS)
//  - O accumulate: each wave owns 32 d-rows x 64 q-cols (8 C tiles, fp32 regs)
// ---------------------------------------------------------------------------
__global__ __launch_bounds__(256) void attn_flash_kernel(
    const __bf16* __restrict__ Kfrag, const __bf16* __restrict__ Qfrag,
    const __bf16* __restrict__ Vfrag, const float* __restrict__ pos,
    float* __restrict__ out)
{
    __shared__ __align__(16) float  Sbuf[64][65];          // padded: no bank conflicts
    __shared__ __align__(32) __bf16 pfrag[2][4][32][16];   // P~ in B-operand layout
    __shared__ __align__(32) __bf16 Qlds[8][4][32][16];    // staged Q fragments
    __shared__ float pm[4][64], pl[4][64];
    __shared__ float mArr[64], lArr[64], cArr[64];

    int blk = blockIdx.x;
    int qt  = blk & 63;
    int n   = blk >> 6;
    int j0  = qt * 64;

    int tid  = threadIdx.x;
    int lane = tid & 31;
    int wv   = tid >> 5;
    int m    = lane & 15;
    int hi   = lane >> 4;

    const v16bf* kf = (const v16bf*)Kfrag;
    const v16bf* vf = (const v16bf*)Vfrag;

    // ---- stage this block's Q fragments (32 KB) into LDS via async-to-LDS ----
    // Linear copy: 8 dc-chunks x 4 KB, as 2048 16B transfers, 8 per thread.
    {
        unsigned lds_base = (unsigned)(uintptr_t)(&Qlds[0][0][0][0]);
        const char* gq = (const char*)(Qfrag + (((size_t)n * 8) * 256 + (j0 >> 4)) * 512);
        // Global layout per dc: 4 contiguous jt-tiles (4 KB), dc-chunks strided 256 tiles.
        for (int c = tid; c < 2048; c += 256) {
            int dc  = c >> 8;          // 256 x 16B per dc-chunk
            int off = c & 255;
            unsigned lds_addr = lds_base + (unsigned)(dc * 4096 + off * 16);
            const char* gaddr = gq + (size_t)dc * (256 * 512 * 2) + (size_t)off * 16;
            asm volatile("global_load_async_to_lds_b128 %0, %1, off"
                         :: "v"(lds_addr), "v"(gaddr)
                         : "memory");
        }
        asm volatile("s_wait_asynccnt 0" ::: "memory");
    }
    if (tid < 64) { mArr[tid] = -3.4e38f; lArr[tid] = 0.f; }

    v8f o_acc[2][4] = {};   // wave owns d-rows [wv*32, wv*32+32) x all 4 j-tiles

    int it  = wv & 3;
    int jt0 = (wv >> 2) * 2;

    for (int kt = 0; kt < 64; ++kt) {     // 64 key tiles of 64
        __syncthreads();                  // prev iter's LDS consumers done
                                          // (also publishes async Q stage on kt==0)

        // ---- S = K^T Q for this wave's two 16x16 tiles ----
#pragma unroll
        for (int t = 0; t < 2; ++t) {
            int jt = jt0 + t;
            v8f s = {0.f, 0.f, 0.f, 0.f, 0.f, 0.f, 0.f, 0.f};
#pragma unroll
            for (int dc = 0; dc < 8; ++dc) {
                v16bf a  = kf[(((size_t)n * 8 + dc) * 256 + (kt * 4 + it)) * 32 + lane];
                v16bf bq = *(const v16bf*)&Qlds[dc][jt][lane][0];
                s = __builtin_amdgcn_wmma_f32_16x16x32_bf16(
                    false, a, false, bq, (short)0, s, false, false);
            }
            int iB = it * 16 + hi * 8;
            int jA = jt * 16 + m;
            // prefetch next key tile's positional-bias lines into cache
            if (kt + 1 < 64)
                __builtin_prefetch(&pos[(size_t)((kt + 1) * 64 + iB) * P_ + j0 + jA], 0, 0);
#pragma unroll
            for (int r = 0; r < 8; ++r) {
                int i_loc = iB + r;
                float pv = pos[(size_t)(kt * 64 + i_loc) * P_ + j0 + jA];
                Sbuf[i_loc][jA] = (s[r] + pv) * 0.0625f;   // (S+pos)/sqrt(256)
            }
        }
        __syncthreads();

        // ---- column max (softmax over key axis i, per query column j) ----
        {
            int j = tid & 63, seg = tid >> 6;
            float mx = -3.4e38f;
#pragma unroll
            for (int r = 0; r < 16; ++r) mx = fmaxf(mx, Sbuf[seg * 16 + r][j]);
            pm[seg][j] = mx;
        }
        __syncthreads();
        if (tid < 64) {
            float mx = fmaxf(fmaxf(pm[0][tid], pm[1][tid]),
                             fmaxf(pm[2][tid], pm[3][tid]));
            float mo = mArr[tid];
            float mn = fmaxf(mo, mx);
            mArr[tid] = mn;
            cArr[tid] = __expf(mo - mn);
        }
        __syncthreads();

        // ---- exp, write P~ (bf16) in B-operand layout, partial sums ----
        {
            int j = tid & 63, seg = tid >> 6;
            float mn = mArr[j];
            float sum = 0.f;
#pragma unroll
            for (int r = 0; r < 16; ++r) {
                int i = seg * 16 + r;
                float e = __expf(Sbuf[i][j] - mn);
                sum += e;
                int kc = i >> 5, il = i & 31;
                pfrag[kc][j >> 4][(j & 15) + (((il & 15) >= 8) ? 16 : 0)][(il >> 4) * 8 + (il & 7)] = f2bf(e);
            }
            pl[seg][j] = sum;
        }
        __syncthreads();
        if (tid < 64)
            lArr[tid] = lArr[tid] * cArr[tid] +
                        (pl[0][tid] + pl[1][tid] + pl[2][tid] + pl[3][tid]);

        // ---- rescale O by exp(m_old - m_new), then O += V * P~ ----
#pragma unroll
        for (int tm = 0; tm < 2; ++tm) {
            int mt = wv * 2 + tm;
#pragma unroll
            for (int jt = 0; jt < 4; ++jt) {
                float c = cArr[jt * 16 + m];
                v8f o = o_acc[tm][jt];
#pragma unroll
                for (int r = 0; r < 8; ++r) o[r] *= c;
#pragma unroll
                for (int kc = 0; kc < 2; ++kc) {
                    v16bf av = vf[(((size_t)n * 128 + (kt * 2 + kc)) * 16 + mt) * 32 + lane];
                    v16bf bp = *(const v16bf*)&pfrag[kc][jt][lane][0];
                    o = __builtin_amdgcn_wmma_f32_16x16x32_bf16(
                        false, av, false, bp, (short)0, o, false, false);
                }
                o_acc[tm][jt] = o;
            }
        }
    }
    __syncthreads();

    // ---- epilogue: divide by l, store out[n, d, j] ----
#pragma unroll
    for (int tm = 0; tm < 2; ++tm) {
        int mt = wv * 2 + tm;
#pragma unroll
        for (int jt = 0; jt < 4; ++jt) {
            float linv = 1.f / lArr[jt * 16 + m];
#pragma unroll
            for (int r = 0; r < 8; ++r) {
                int d_abs = mt * 16 + r + hi * 8;
                out[((size_t)n * D_ + d_abs) * P_ + j0 + jt * 16 + m] =
                    o_acc[tm][jt][r] * linv;
            }
        }
    }
}

extern "C" void kernel_launch(void* const* d_in, const int* in_sizes, int n_in,
                              void* d_out, int out_size, void* d_ws, size_t ws_size,
                              hipStream_t stream) {
    const float* x   = (const float*)d_in[0];
    const float* pos = (const float*)d_in[1];
    const float* Wk  = (const float*)d_in[2];
    const float* bk  = (const float*)d_in[3];
    const float* Wq  = (const float*)d_in[4];
    const float* bq  = (const float*)d_in[5];
    const float* Wv  = (const float*)d_in[6];
    const float* bv  = (const float*)d_in[7];
    float* out = (float*)d_out;

    char* ws = (char*)d_ws;
    const size_t FRAG_BYTES = (size_t)N_ * D_ * P_ * 2;   // 8 MB each
    __bf16* Kfrag = (__bf16*)(ws);
    __bf16* Qfrag = (__bf16*)(ws + FRAG_BYTES);
    __bf16* Vfrag = (__bf16*)(ws + 2 * FRAG_BYTES);

    // Phase 1: 3 matrices x 4 batches x 16 o-tiles x 32 p-groups
    proj_frag_kernel<<<3 * 4 * 16 * 32, 256, 0, stream>>>(
        x, Wk, bk, Wq, bq, Wv, bv, Kfrag, Qfrag, Vfrag);

    // Phase 2: 4 batches x 64 query tiles of 64
    attn_flash_kernel<<<N_ * (P_ / 64), 256, 0, stream>>>(
        Kfrag, Qfrag, Vfrag, pos, out);
}
